// Encoder_26972394619576
// MI455X (gfx1250) — matrix operand order repaired and checked by reading, not compiled, not run on hardware
//
#include <hip/hip_runtime.h>

// ---------------------------------------------------------------------------
// Transformer encoder layer for MI455X (gfx1250, wave32, WMMA).
// B=4, S=2048, D=512, F=2048. All GEMMs run on v_wmma_f32_16x16x32_bf16.
// GEMMs consume B as [N,K] row-major ("BT" form) so LDS staging is b128-only.
// Global->LDS staging is double-buffered through registers so HBM latency is
// hidden behind the WMMA block (one s_wait_loadcnt per K-step).
// ---------------------------------------------------------------------------

typedef __attribute__((ext_vector_type(16))) __bf16 v16bf;
typedef __attribute__((ext_vector_type(8)))  float  v8f;

union FragBF {
    v16bf v;
    uint4 u[2];
};

static constexpr int kB  = 4;
static constexpr int kS  = 2048;
static constexpr int kD  = 512;
static constexpr int kF  = 2048;
static constexpr int kBS = kB * kS;       // 8192 rows
static constexpr float kEPS = 1e-5f;

__device__ __forceinline__ unsigned short f32_to_bf16(float f) {
    unsigned int u = __float_as_uint(f);
    unsigned int r = (u + 0x7FFFu + ((u >> 16) & 1u)) >> 16;  // RNE
    return (unsigned short)r;
}
__device__ __forceinline__ float bf16_to_f32(unsigned short h) {
    return __uint_as_float(((unsigned int)h) << 16);
}

// ---------------------------------------------------------------------------
// Elementwise f32 -> bf16 conversion (vectorized: float4 -> 4x bf16)
// ---------------------------------------------------------------------------
__global__ void cvt_f32_bf16(const float4* __restrict__ in,
                             ushort2* __restrict__ out, long n4) {
    long i = (long)blockIdx.x * blockDim.x + threadIdx.x;
    if (i < n4) {
        float4 f = in[i];
        out[2 * i + 0] = make_ushort2(f32_to_bf16(f.x), f32_to_bf16(f.y));
        out[2 * i + 1] = make_ushort2(f32_to_bf16(f.z), f32_to_bf16(f.w));
    }
}

// ---------------------------------------------------------------------------
// Fused transpose + f32->bf16:  out[c][r] = (bf16) in[r][c]
// in: [R,C] f32 row-major; out: [C,R] bf16 row-major. R,C multiples of 32.
// ---------------------------------------------------------------------------
__global__ __launch_bounds__(256)
void transpose_f32_bf16(const float* __restrict__ in,
                        unsigned short* __restrict__ out, int R, int C) {
    __shared__ unsigned short tile[32][33];
    const int bx = blockIdx.x * 32;   // C offset
    const int by = blockIdx.y * 32;   // R offset
    const int tx = threadIdx.x & 31;
    const int ty = threadIdx.x >> 5;  // 0..7
    #pragma unroll
    for (int i = 0; i < 32; i += 8)
        tile[ty + i][tx] = f32_to_bf16(in[(size_t)(by + ty + i) * C + bx + tx]);
    __syncthreads();
    #pragma unroll
    for (int i = 0; i < 32; i += 8)
        out[(size_t)(bx + ty + i) * R + by + tx] = tile[tx][ty + i];
}

// ---------------------------------------------------------------------------
// bf16 WMMA GEMM, BT form:  C[M,N] = scale * (A[M,K] x Bt[N,K]^T) + bias.
// Workgroup tile 128(M) x 128(N), K-step 32, 8 waves (4x2), each wave a
// 32x64 patch = 2x4 fragments of v_wmma_f32_16x16x32_bf16.
// Double-buffered LDS staging: the next K-tile is loaded into registers
// while the current tile is consumed from LDS.
// blockIdx.z is batch; sA/sB/sC are per-batch element offsets; lda/ldb/ldc
// are row strides in elements (ldc = N normally, M when OUTTRANS).
// ---------------------------------------------------------------------------
template <bool RELU, bool OUTBF16, bool OUTTRANS>
__global__ __launch_bounds__(256)
void gemm_bt_wmma(const unsigned short* __restrict__ A, size_t sA, int lda,
                  const unsigned short* __restrict__ Bt, size_t sB, int ldb,
                  void* __restrict__ C, size_t sC, int ldc,
                  const float* __restrict__ bias,
                  int M, int N, int K, float scale) {
    __shared__ unsigned short As[2][128][32];   // [buf][m][k]
    __shared__ unsigned short Bs[2][128][32];   // [buf][n][k]

    const int zb = blockIdx.z;
    A  += (size_t)zb * sA;
    Bt += (size_t)zb * sB;

    const int t      = threadIdx.x;
    const int mBlock = blockIdx.y * 128;
    const int nBlock = blockIdx.x * 128;

    const int lane = t & 31;
    const int w    = t >> 5;            // wave id 0..7
    const int wm   = (w >> 1) * 32;     // 0,32,64,96
    const int wn   = (w & 1)  * 64;     // 0,64
    const int lr   = lane & 15;
    const int hi   = lane >> 4;         // 0 or 1

    // Staging: 128 rows x 32 cols = 512 uint4 per operand; this thread covers
    // rows srow0 and srow0+64, chunk sc4.
    const int srow0 = t >> 2;
    const int sc4   = t & 3;

    const unsigned short* aP0 = A  + (size_t)(mBlock + srow0)      * lda + sc4 * 8;
    const unsigned short* aP1 = A  + (size_t)(mBlock + srow0 + 64) * lda + sc4 * 8;
    const unsigned short* bP0 = Bt + (size_t)(nBlock + srow0)      * ldb + sc4 * 8;
    const unsigned short* bP1 = Bt + (size_t)(nBlock + srow0 + 64) * ldb + sc4 * 8;

    uint4 ra0, ra1, rb0, rb1;

    // prologue: tile 0 -> buffer 0
    ra0 = *(const uint4*)(aP0);
    ra1 = *(const uint4*)(aP1);
    rb0 = *(const uint4*)(bP0);
    rb1 = *(const uint4*)(bP1);
    *(uint4*)(&As[0][srow0][sc4 * 8])      = ra0;
    *(uint4*)(&As[0][srow0 + 64][sc4 * 8]) = ra1;
    *(uint4*)(&Bs[0][srow0][sc4 * 8])      = rb0;
    *(uint4*)(&Bs[0][srow0 + 64][sc4 * 8]) = rb1;
    __syncthreads();

    const v8f vzero = {0.f, 0.f, 0.f, 0.f, 0.f, 0.f, 0.f, 0.f};
    v8f acc[2][4];
    #pragma unroll
    for (int i = 0; i < 2; ++i)
        #pragma unroll
        for (int j = 0; j < 4; ++j) acc[i][j] = vzero;

    int buf = 0;
    for (int k0 = 0; k0 < K; k0 += 32) {
        const bool more = (k0 + 32) < K;
        // ---- issue next tile's global loads (latency hidden by WMMAs) ----
        if (more) {
            ra0 = *(const uint4*)(aP0 + k0 + 32);
            ra1 = *(const uint4*)(aP1 + k0 + 32);
            rb0 = *(const uint4*)(bP0 + k0 + 32);
            rb1 = *(const uint4*)(bP1 + k0 + 32);
        }
        if (k0 + 64 < K) {
            __builtin_prefetch(aP0 + k0 + 64, 0, 1);
            __builtin_prefetch(bP0 + k0 + 64, 0, 1);
        }

        // ---- fragments per ISA 16-bit A/B VGPR layouts ----
        FragBF fa[2], fb[4];
        #pragma unroll
        for (int fm = 0; fm < 2; ++fm) {
            const unsigned short* ap = &As[buf][wm + fm * 16 + lr][0];
            fa[fm].u[0] = *(const uint4*)(ap + hi * 8);        // K = hi*8 .. +7
            fa[fm].u[1] = *(const uint4*)(ap + 16 + hi * 8);   // K = 16+hi*8 .. +7
        }
        #pragma unroll
        for (int fn = 0; fn < 4; ++fn) {
            const unsigned short* bp = &Bs[buf][wn + fn * 16 + lr][0];
            fb[fn].u[0] = *(const uint4*)(bp + hi * 8);        // K = hi*8 .. +7
            fb[fn].u[1] = *(const uint4*)(bp + 16 + hi * 8);   // K = 16+hi*8 .. +7
        }
        #pragma unroll
        for (int fm = 0; fm < 2; ++fm)
            #pragma unroll
            for (int fn = 0; fn < 4; ++fn)
                acc[fm][fn] = __builtin_amdgcn_wmma_f32_16x16x32_bf16(
                    false, fa[fm].v, false, fb[fn].v,
                    (short)0, acc[fm][fn], false, false);

        // ---- drain staged registers into the other buffer ----
        if (more) {
            const int nb = buf ^ 1;
            *(uint4*)(&As[nb][srow0][sc4 * 8])      = ra0;
            *(uint4*)(&As[nb][srow0 + 64][sc4 * 8]) = ra1;
            *(uint4*)(&Bs[nb][srow0][sc4 * 8])      = rb0;
            *(uint4*)(&Bs[nb][srow0 + 64][sc4 * 8]) = rb1;
            __syncthreads();
            buf = nb;
        }
    }

    // ---- epilogue: C layout — VGPR r: M = r + hi*8, N = lane%16 ----
    float* outf = (float*)C + (size_t)zb * sC;
    unsigned short* outb = (unsigned short*)C + (size_t)zb * sC;
    #pragma unroll
    for (int fm = 0; fm < 2; ++fm) {
        #pragma unroll
        for (int fn = 0; fn < 4; ++fn) {
            int col  = nBlock + wn + fn * 16 + lr;
            float bv = bias ? bias[col] : 0.f;
            #pragma unroll
            for (int r = 0; r < 8; ++r) {
                int rowg = mBlock + wm + fm * 16 + hi * 8 + r;
                float vv = acc[fm][fn][r] * scale + bv;
                if (RELU) vv = fmaxf(vv, 0.f);
                size_t o = OUTTRANS ? ((size_t)col * ldc + rowg)
                                    : ((size_t)rowg * ldc + col);
                if (OUTBF16) outb[o] = f32_to_bf16(vv);
                else         outf[o] = vv;
            }
        }
    }
}

// ---------------------------------------------------------------------------
// Row softmax over 2048 bf16 values, in place. One block (256 thr) per row;
// each thread handles one 8-element (uint4) chunk.
// ---------------------------------------------------------------------------
__global__ __launch_bounds__(256)
void softmax_rows(unsigned short* __restrict__ p) {
    __shared__ float sm[256];
    const int t = threadIdx.x;
    uint4* row4 = (uint4*)(p + (size_t)blockIdx.x * kS);

    uint4 raw = row4[t];
    unsigned short* pv = (unsigned short*)&raw;
    float x[8];
    float lmax = -3.4e38f;
    #pragma unroll
    for (int i = 0; i < 8; ++i) {
        x[i] = bf16_to_f32(pv[i]);
        lmax = fmaxf(lmax, x[i]);
    }
    sm[t] = lmax; __syncthreads();
    for (int o = 128; o > 0; o >>= 1) {
        if (t < o) sm[t] = fmaxf(sm[t], sm[t + o]);
        __syncthreads();
    }
    float rmax = sm[0]; __syncthreads();

    float lsum = 0.f;
    #pragma unroll
    for (int i = 0; i < 8; ++i) {
        x[i] = __expf(x[i] - rmax);
        lsum += x[i];
    }
    sm[t] = lsum; __syncthreads();
    for (int o = 128; o > 0; o >>= 1) {
        if (t < o) sm[t] += sm[t + o];
        __syncthreads();
    }
    float inv = 1.0f / sm[0];
    #pragma unroll
    for (int i = 0; i < 8; ++i)
        pv[i] = f32_to_bf16(x[i] * inv);
    row4[t] = raw;
}

// ---------------------------------------------------------------------------
// Residual add + LayerNorm over rows of 512. Writes f32 out and optional
// bf16 copy. One block (256 thr) per row; float2 per thread.
// ---------------------------------------------------------------------------
__global__ __launch_bounds__(256)
void add_layernorm(const float* __restrict__ a, const float* __restrict__ b,
                   const float* __restrict__ g, const float* __restrict__ be,
                   float* __restrict__ outf, unsigned short* __restrict__ outb) {
    __shared__ float sm[256];
    const int t = threadIdx.x;
    const size_t base2 = (size_t)blockIdx.x * (kD / 2) + t;   // float2 index

    float2 av = ((const float2*)a)[base2];
    float2 bv = ((const float2*)b)[base2];
    float s0 = av.x + bv.x;
    float s1 = av.y + bv.y;

    sm[t] = s0 + s1; __syncthreads();
    for (int o = 128; o > 0; o >>= 1) {
        if (t < o) sm[t] += sm[t + o];
        __syncthreads();
    }
    float mu = sm[0] * (1.0f / kD); __syncthreads();

    float d0 = s0 - mu, d1 = s1 - mu;
    sm[t] = d0 * d0 + d1 * d1; __syncthreads();
    for (int o = 128; o > 0; o >>= 1) {
        if (t < o) sm[t] += sm[t + o];
        __syncthreads();
    }
    float rstd = rsqrtf(sm[0] * (1.0f / kD) + kEPS);

    float2 gv = ((const float2*)g)[t];
    float2 bev = ((const float2*)be)[t];
    float y0 = d0 * rstd * gv.x + bev.x;
    float y1 = d1 * rstd * gv.y + bev.y;
    ((float2*)outf)[base2] = make_float2(y0, y1);
    if (outb)
        ((ushort2*)outb)[base2] = make_ushort2(f32_to_bf16(y0), f32_to_bf16(y1));
}

// ---------------------------------------------------------------------------
// Host orchestration
// ---------------------------------------------------------------------------
extern "C" void kernel_launch(void* const* d_in, const int* in_sizes, int n_in,
                              void* d_out, int out_size, void* d_ws, size_t ws_size,
                              hipStream_t stream) {
    (void)in_sizes; (void)n_in; (void)out_size; (void)ws_size;

    const float* q   = (const float*)d_in[0];
    const float* k   = (const float*)d_in[1];
    const float* v   = (const float*)d_in[2];
    const float* x   = (const float*)d_in[3];
    const float* Wq  = (const float*)d_in[4];
    const float* bq  = (const float*)d_in[5];
    const float* Wk  = (const float*)d_in[6];
    const float* bk  = (const float*)d_in[7];
    const float* Wv  = (const float*)d_in[8];
    const float* bv  = (const float*)d_in[9];
    const float* g1  = (const float*)d_in[10];
    const float* be1 = (const float*)d_in[11];
    const float* W1  = (const float*)d_in[12];
    const float* b1  = (const float*)d_in[13];
    const float* W2  = (const float*)d_in[14];
    const float* b2  = (const float*)d_in[15];
    const float* g2  = (const float*)d_in[16];
    const float* be2 = (const float*)d_in[17];
    float* out = (float*)d_out;

    // Workspace carve-up (256B aligned)
    char* base = (char*)d_ws;
    size_t off = 0;
    auto alloc = [&](size_t bytes) -> char* {
        char* p = base + off;
        off = (off + bytes + 255) & ~(size_t)255;
        return p;
    };
    unsigned short* wqT = (unsigned short*)alloc((size_t)kD * kD * 2);  // [D,D] = Wq^T
    unsigned short* wkT = (unsigned short*)alloc((size_t)kD * kD * 2);
    unsigned short* wvT = (unsigned short*)alloc((size_t)kD * kD * 2);
    unsigned short* w1T = (unsigned short*)alloc((size_t)kF * kD * 2);  // [F,D] = W1^T
    unsigned short* w2T = (unsigned short*)alloc((size_t)kD * kF * 2);  // [D,F] = W2^T
    unsigned short* qb  = (unsigned short*)alloc((size_t)kBS * kD * 2);
    unsigned short* kb  = (unsigned short*)alloc((size_t)kBS * kD * 2);
    unsigned short* vb  = (unsigned short*)alloc((size_t)kBS * kD * 2);
    unsigned short* Qb  = (unsigned short*)alloc((size_t)kBS * kD * 2);
    unsigned short* Kb  = (unsigned short*)alloc((size_t)kBS * kD * 2);
    unsigned short* VbT = (unsigned short*)alloc((size_t)kD * kBS * 2); // [D, B*S]
    unsigned short* sc  = (unsigned short*)alloc((size_t)kB * kS * kS * 2);
    float*          at  = (float*)alloc((size_t)kBS * kD * 4);
    float*          hf  = (float*)alloc((size_t)kBS * kD * 4);
    unsigned short* hb  = (unsigned short*)alloc((size_t)kBS * kD * 2);
    unsigned short* fbuf= (unsigned short*)alloc((size_t)kBS * kF * 2);
    float*          f2  = (float*)alloc((size_t)kBS * kD * 4);

    const dim3 blk(256);

    // Weight transposes (fused f32->bf16): out[c][r] = in[r][c]
    transpose_f32_bf16<<<dim3(kD / 32, kD / 32), blk, 0, stream>>>(Wq, wqT, kD, kD);
    transpose_f32_bf16<<<dim3(kD / 32, kD / 32), blk, 0, stream>>>(Wk, wkT, kD, kD);
    transpose_f32_bf16<<<dim3(kD / 32, kD / 32), blk, 0, stream>>>(Wv, wvT, kD, kD);
    transpose_f32_bf16<<<dim3(kF / 32, kD / 32), blk, 0, stream>>>(W1, w1T, kD, kF);
    transpose_f32_bf16<<<dim3(kD / 32, kF / 32), blk, 0, stream>>>(W2, w2T, kF, kD);

    // Activation conversions (vectorized)
    auto cvt = [&](const float* src, unsigned short* dst, long n) {
        long n4 = n / 4;
        cvt_f32_bf16<<<dim3((unsigned)((n4 + 255) / 256)), blk, 0, stream>>>(
            (const float4*)src, (ushort2*)dst, n4);
    };
    cvt(q, qb, (long)kBS * kD);
    cvt(k, kb, (long)kBS * kD);
    cvt(v, vb, (long)kBS * kD);

    // QKV projections: [8192,512] x [512,512]^T(stored [N,K]) -> bf16
    gemm_bt_wmma<false, true, false><<<dim3(kD / 128, kBS / 128, 1), blk, 0, stream>>>(
        qb, 0, kD, wqT, 0, kD, Qb, 0, kD, bq, kBS, kD, kD, 1.0f);
    gemm_bt_wmma<false, true, false><<<dim3(kD / 128, kBS / 128, 1), blk, 0, stream>>>(
        kb, 0, kD, wkT, 0, kD, Kb, 0, kD, bk, kBS, kD, kD, 1.0f);
    // V projection writes V^T [D, B*S] directly (B operand for attn GEMM)
    gemm_bt_wmma<false, true, true><<<dim3(kD / 128, kBS / 128, 1), blk, 0, stream>>>(
        vb, 0, kD, wvT, 0, kD, VbT, 0, kBS, bv, kBS, kD, kD, 1.0f);

    // scores = Q x K^T / sqrt(D): per batch [2048,512] x [2048,512]^T -> bf16
    const float invsqrtd = 0.044194173824159216f;  // 1/sqrt(512)
    gemm_bt_wmma<false, true, false><<<dim3(kS / 128, kS / 128, kB), blk, 0, stream>>>(
        Qb, (size_t)kS * kD, kD, Kb, (size_t)kS * kD, kD, sc, (size_t)kS * kS, kS,
        nullptr, kS, kS, kD, invsqrtd);

    // softmax over each row of 2048
    softmax_rows<<<dim3(kBS), blk, 0, stream>>>(sc);

    // attn = P x V: per batch [2048,2048] x V^T[512 rows, 2048 col window] -> f32
    gemm_bt_wmma<false, false, false><<<dim3(kD / 128, kS / 128, kB), blk, 0, stream>>>(
        sc, (size_t)kS * kS, kS, VbT, (size_t)kS /*col window*/, kBS,
        at, (size_t)kS * kD, kD, nullptr, kS, kD, kS, 1.0f);

    // h = LayerNorm(attn + x)
    add_layernorm<<<dim3(kBS), blk, 0, stream>>>(at, x, g1, be1, hf, hb);

    // FFN1: relu([8192,512] x W1^T[2048,512] + b1) -> bf16
    gemm_bt_wmma<true, true, false><<<dim3(kF / 128, kBS / 128, 1), blk, 0, stream>>>(
        hb, 0, kD, w1T, 0, kD, fbuf, 0, kF, b1, kBS, kF, kD, 1.0f);

    // FFN2: [8192,2048] x W2^T[512,2048] + b2 -> f32
    gemm_bt_wmma<false, false, false><<<dim3(kD / 128, kBS / 128, 1), blk, 0, stream>>>(
        fbuf, 0, kF, w2T, 0, kF, f2, 0, kD, b2, kBS, kD, kF, 1.0f);

    // out = LayerNorm(f2 + h)
    add_layernorm<<<dim3(kBS), blk, 0, stream>>>(f2, hf, g2, be2, out, nullptr);
}
